// InvariantParticleAttention_26156350832784
// MI455X (gfx1250) — matrix operand order repaired and checked by reading, not compiled, not run on hardware
//
#include <hip/hip_runtime.h>
#include <hip/hip_bf16.h>

// ---------------------------------------------------------------------------
// InvariantParticleAttention for MI455X (gfx1250, wave32, WMMA)
//   B=32, H=8, N=512, C=64 (16 scalars + 12 Lorentz 4-vectors)
// ---------------------------------------------------------------------------

typedef __bf16 v16bf __attribute__((ext_vector_type(16)));
typedef float  v8f   __attribute__((ext_vector_type(8)));

union AF { v16bf v; uint4 u[2]; };   // 8 VGPRs = 16 bf16 per lane

constexpr int BB  = 32;
constexpr int HH  = 8;
constexpr int NN  = 512;
constexpr int CC  = 64;
constexpr int NSC = 16;   // scalar channels
constexpr int NVEC = 12;  // 4-vector channels

// Async global->LDS (CDNA5 GLOBAL_LOAD_ASYNC_TO_LDS_B128, ASYNCcnt tracked).
// Probe-confirmed prototype: (v4i addrspace(1)*, v4i addrspace(3)*, imm, imm).
#if __has_builtin(__builtin_amdgcn_global_load_async_to_lds_b128) && \
    __has_builtin(__builtin_amdgcn_s_wait_asynccnt)
#define IPA_ASYNC_LDS 1
typedef int ipa_v4i __attribute__((ext_vector_type(4)));
typedef __attribute__((address_space(1))) ipa_v4i* ipa_gv4p;
typedef __attribute__((address_space(3))) ipa_v4i* ipa_lv4p;
#else
#define IPA_ASYNC_LDS 0
#endif

// 16-lane XOR butterfly exchange. ds_swizzle group-of-32 mode:
// new_lane = ((lane & and) | or) ^ xor, offset = {xor[14:10], or[9:5], and[4:0]}.
template <int MASK>
__device__ __forceinline__ float lane_xor(float x) {
#if __has_builtin(__builtin_amdgcn_ds_swizzle)
  return __builtin_bit_cast(
      float, __builtin_amdgcn_ds_swizzle(__builtin_bit_cast(int, x),
                                         0x1f | (MASK << 10)));
#else
  return __shfl_xor(x, MASK, 32);
#endif
}

// ---------------------------------------------------------------------------
// Kernel 1: per-(b,n) 4x4 inverse and eta*inv*eta (adjugate method)
// ---------------------------------------------------------------------------
__global__ __launch_bounds__(256) void ipa_frames(const float* __restrict__ lf,
                                                  float* __restrict__ invA,
                                                  float* __restrict__ invL) {
  const size_t idx = (size_t)blockIdx.x * 256 + threadIdx.x;  // < B*N
  const float* m = lf + idx * 16;
  float mm[16];
#pragma unroll
  for (int i = 0; i < 16; ++i) mm[i] = m[i];

  float inv[16];
  inv[0]  =  mm[5]*mm[10]*mm[15] - mm[5]*mm[11]*mm[14] - mm[9]*mm[6]*mm[15] + mm[9]*mm[7]*mm[14] + mm[13]*mm[6]*mm[11] - mm[13]*mm[7]*mm[10];
  inv[4]  = -mm[4]*mm[10]*mm[15] + mm[4]*mm[11]*mm[14] + mm[8]*mm[6]*mm[15] - mm[8]*mm[7]*mm[14] - mm[12]*mm[6]*mm[11] + mm[12]*mm[7]*mm[10];
  inv[8]  =  mm[4]*mm[9]*mm[15]  - mm[4]*mm[11]*mm[13] - mm[8]*mm[5]*mm[15] + mm[8]*mm[7]*mm[13] + mm[12]*mm[5]*mm[11] - mm[12]*mm[7]*mm[9];
  inv[12] = -mm[4]*mm[9]*mm[14]  + mm[4]*mm[10]*mm[13] + mm[8]*mm[5]*mm[14] - mm[8]*mm[6]*mm[13] - mm[12]*mm[5]*mm[10] + mm[12]*mm[6]*mm[9];
  inv[1]  = -mm[1]*mm[10]*mm[15] + mm[1]*mm[11]*mm[14] + mm[9]*mm[2]*mm[15] - mm[9]*mm[3]*mm[14] - mm[13]*mm[2]*mm[11] + mm[13]*mm[3]*mm[10];
  inv[5]  =  mm[0]*mm[10]*mm[15] - mm[0]*mm[11]*mm[14] - mm[8]*mm[2]*mm[15] + mm[8]*mm[3]*mm[14] + mm[12]*mm[2]*mm[11] - mm[12]*mm[3]*mm[10];
  inv[9]  = -mm[0]*mm[9]*mm[15]  + mm[0]*mm[11]*mm[13] + mm[8]*mm[1]*mm[15] - mm[8]*mm[3]*mm[13] - mm[12]*mm[1]*mm[11] + mm[12]*mm[3]*mm[9];
  inv[13] =  mm[0]*mm[9]*mm[14]  - mm[0]*mm[10]*mm[13] - mm[8]*mm[1]*mm[14] + mm[8]*mm[2]*mm[13] + mm[12]*mm[1]*mm[10] - mm[12]*mm[2]*mm[9];
  inv[2]  =  mm[1]*mm[6]*mm[15]  - mm[1]*mm[7]*mm[14]  - mm[5]*mm[2]*mm[15] + mm[5]*mm[3]*mm[14] + mm[13]*mm[2]*mm[7]  - mm[13]*mm[3]*mm[6];
  inv[6]  = -mm[0]*mm[6]*mm[15]  + mm[0]*mm[7]*mm[14]  + mm[4]*mm[2]*mm[15] - mm[4]*mm[3]*mm[14] - mm[12]*mm[2]*mm[7]  + mm[12]*mm[3]*mm[6];
  inv[10] =  mm[0]*mm[5]*mm[15]  - mm[0]*mm[7]*mm[13]  - mm[4]*mm[1]*mm[15] + mm[4]*mm[3]*mm[13] + mm[12]*mm[1]*mm[7]  - mm[12]*mm[3]*mm[5];
  inv[14] = -mm[0]*mm[5]*mm[14]  + mm[0]*mm[6]*mm[13]  + mm[4]*mm[1]*mm[14] - mm[4]*mm[2]*mm[13] - mm[12]*mm[1]*mm[6]  + mm[12]*mm[2]*mm[5];
  inv[3]  = -mm[1]*mm[6]*mm[11]  + mm[1]*mm[7]*mm[10]  + mm[5]*mm[2]*mm[11] - mm[5]*mm[3]*mm[10] - mm[9]*mm[2]*mm[7]   + mm[9]*mm[3]*mm[6];
  inv[7]  =  mm[0]*mm[6]*mm[11]  - mm[0]*mm[7]*mm[10]  - mm[4]*mm[2]*mm[11] + mm[4]*mm[3]*mm[10] + mm[8]*mm[2]*mm[7]   - mm[8]*mm[3]*mm[6];
  inv[11] = -mm[0]*mm[5]*mm[11]  + mm[0]*mm[7]*mm[9]   + mm[4]*mm[1]*mm[11] - mm[4]*mm[3]*mm[9]  - mm[8]*mm[1]*mm[7]   + mm[8]*mm[3]*mm[5];
  inv[15] =  mm[0]*mm[5]*mm[10]  - mm[0]*mm[6]*mm[9]   - mm[4]*mm[1]*mm[10] + mm[4]*mm[2]*mm[9]  + mm[8]*mm[1]*mm[6]   - mm[8]*mm[2]*mm[5];

  const float det = mm[0]*inv[0] + mm[1]*inv[4] + mm[2]*inv[8] + mm[3]*inv[12];
  const float d = 1.0f / det;

  float* oA = invA + idx * 16;
  float* oL = invL + idx * 16;
#pragma unroll
  for (int i = 0; i < 4; ++i) {
    const float si = (i == 0) ? 1.0f : -1.0f;
#pragma unroll
    for (int j = 0; j < 4; ++j) {
      const float sj = (j == 0) ? 1.0f : -1.0f;
      const float x = inv[i * 4 + j] * d;
      oA[i * 4 + j] = x;
      oL[i * 4 + j] = si * sj * x;
    }
  }
}

// ---------------------------------------------------------------------------
// Kernel 2: rotate q/k/v channel rows, emit bf16 tensors.
//   Qg     : [B*H, N, C]  pre-scaled by 1/sqrt(C)
//   Kg     : [B*H, N, C]
//   Vt     : [B*H, C, N]  (channel major, WMMA B-of-PV friendly)
// ---------------------------------------------------------------------------
__device__ __forceinline__ void rot_row(const float* __restrict__ r,
                                        const float* __restrict__ M,
                                        float* __restrict__ o) {
#pragma unroll
  for (int c = 0; c < NSC; ++c) o[c] = r[c];
#pragma unroll
  for (int vv = 0; vv < NVEC; ++vv) {
    const int base = NSC + vv * 4;
    const float x0 = r[base + 0], x1 = r[base + 1], x2 = r[base + 2], x3 = r[base + 3];
#pragma unroll
    for (int i = 0; i < 4; ++i)
      o[base + i] = M[i * 4 + 0] * x0 + M[i * 4 + 1] * x1 + M[i * 4 + 2] * x2 + M[i * 4 + 3] * x3;
  }
}

__global__ __launch_bounds__(256) void ipa_transform(const float* __restrict__ q,
                                                     const float* __restrict__ k,
                                                     const float* __restrict__ v,
                                                     const float* __restrict__ invA,
                                                     const float* __restrict__ invL,
                                                     __bf16* __restrict__ Qg,
                                                     __bf16* __restrict__ Kg,
                                                     __bf16* __restrict__ Vt) {
  const size_t idx = (size_t)blockIdx.x * 256 + threadIdx.x;  // < B*H*N
  const int n = (int)(idx % NN);
  const size_t bh = idx / NN;
  const int b = (int)(bh / HH);

  const float* Mi = invA + ((size_t)b * NN + n) * 16;
  const float* Ml = invL + ((size_t)b * NN + n) * 16;
  const size_t off = idx * CC;

  float r[CC], o[CC];
  constexpr float kQScale = 0.125f;  // 1/sqrt(C), folded into Q

  // ---- Q (frame = inv), pre-scaled
  {
    const float4* s4 = (const float4*)(q + off);
    float4* r4 = (float4*)r;
#pragma unroll
    for (int i = 0; i < CC / 4; ++i) r4[i] = s4[i];
    rot_row(r, Mi, o);
    __bf16* dst = Qg + off;
#pragma unroll
    for (int c = 0; c < CC; ++c) dst[c] = (__bf16)(o[c] * kQScale);
  }
  // ---- K (frame = eta*inv*eta)
  {
    const float4* s4 = (const float4*)(k + off);
    float4* r4 = (float4*)r;
#pragma unroll
    for (int i = 0; i < CC / 4; ++i) r4[i] = s4[i];
    rot_row(r, Ml, o);
    __bf16* dst = Kg + off;
#pragma unroll
    for (int c = 0; c < CC; ++c) dst[c] = (__bf16)o[c];
  }
  // ---- V (frame = inv), stored transposed [bh, c, n]
  {
    const float4* s4 = (const float4*)(v + off);
    float4* r4 = (float4*)r;
#pragma unroll
    for (int i = 0; i < CC / 4; ++i) r4[i] = s4[i];
    rot_row(r, Mi, o);
    const size_t vtbase = bh * (size_t)CC * NN + n;
#pragma unroll
    for (int c = 0; c < CC; ++c) Vt[vtbase + (size_t)c * NN] = (__bf16)o[c];
  }
}

// ---------------------------------------------------------------------------
// Kernel 3: flash attention. One block = (b,h) x 128 queries. 8 waves,
// each wave owns a 16-query tile. K/V streamed through LDS in 32-key tiles
// via async global->LDS copies; next tile prefetched during compute.
// Per 32-key tile: 4x v_wmma (QK^T) + online softmax (row max via ds_swizzle
// butterflies, row sum via a P*ones WMMA) + 4x v_wmma (PV).
// Epilogue applies lframes rotation and writes fp32 output.
// ---------------------------------------------------------------------------
__global__ __launch_bounds__(256) void ipa_attn(const __bf16* __restrict__ Qg,
                                                const __bf16* __restrict__ Kg,
                                                const __bf16* __restrict__ Vt,
                                                const float* __restrict__ lframes,
                                                float* __restrict__ out) {
  const int tid   = threadIdx.x;
  const int wave  = tid >> 5;
  const int lane  = tid & 31;
  const int l16   = lane & 15;
  const int lhalf = lane >> 4;
  const int bh    = blockIdx.y;
  const int b     = bh >> 3;  // H = 8
  const int q0    = blockIdx.x * 128 + wave * 16;

  __shared__ __bf16 Ksh[32 * CC];       // [key][ch]    4 KB
  __shared__ __bf16 Vsh[CC * 32];       // [ch][key]    4 KB
  __shared__ __bf16 Pbuf[8][16 * 32];   // per-wave P transpose  8 KB
  __shared__ float  Obuf[8][16 * CC];   // per-wave O staging   32 KB

  // --- Q fragments (A layout: lane = row l16; K = lhalf*8+{0..7}, 16+lhalf*8+{0..7})
  const __bf16* Qrow = Qg + ((size_t)bh * NN + q0 + l16) * CC;
  AF qa[2];
#pragma unroll
  for (int kc = 0; kc < 2; ++kc) {
    qa[kc].u[0] = *(const uint4*)(Qrow + kc * 32 + lhalf * 8);
    qa[kc].u[1] = *(const uint4*)(Qrow + kc * 32 + 16 + lhalf * 8);
  }

  // constant all-ones B fragment (bf16 1.0 = 0x3f80): row-sum via WMMA
  AF ones;
  ones.u[0] = make_uint4(0x3f803f80u, 0x3f803f80u, 0x3f803f80u, 0x3f803f80u);
  ones.u[1] = ones.u[0];

  // per-thread source cursors for the cooperative 16 B/thread tile copies
  const __bf16* ksrc = Kg + ((size_t)bh * NN + (tid >> 3)) * CC + (tid & 7) * 8;
  const __bf16* vsrc = Vt + ((size_t)bh * CC + (tid >> 2)) * NN + (tid & 3) * 8;
  __bf16* kdst = Ksh + tid * 8;
  __bf16* vdst = Vsh + tid * 8;

  float mrow[8], lrow[8];
  v8f oacc[4];
#pragma unroll
  for (int j = 0; j < 8; ++j) { mrow[j] = -3.0e38f; lrow[j] = 0.0f; }
#pragma unroll
  for (int t = 0; t < 4; ++t)
#pragma unroll
    for (int j = 0; j < 8; ++j) oacc[t][j] = 0.0f;

  for (int kt0 = 0; kt0 < NN; kt0 += 32) {
    __syncthreads();  // previous tile fully consumed
#if IPA_ASYNC_LDS
    // direct global->LDS, no VGPR round trip (ASYNCcnt tracked)
    __builtin_amdgcn_global_load_async_to_lds_b128(
        (ipa_gv4p)(ksrc + (size_t)kt0 * CC), (ipa_lv4p)kdst, 0, 0);
    __builtin_amdgcn_global_load_async_to_lds_b128(
        (ipa_gv4p)(vsrc + kt0), (ipa_lv4p)vdst, 0, 0);
    __builtin_amdgcn_s_wait_asynccnt(0);
#else
    *(uint4*)kdst = *(const uint4*)(ksrc + (size_t)kt0 * CC);
    *(uint4*)vdst = *(const uint4*)(vsrc + kt0);
#endif
    __syncthreads();  // tile visible to all waves

    // prefetch next tile into cache while we compute on this one
    if (kt0 + 32 < NN) {
      __builtin_prefetch(ksrc + (size_t)(kt0 + 32) * CC, 0, 1);
      __builtin_prefetch(vsrc + kt0 + 32, 0, 1);
    }

    // --- scores S = Q * K^T  (16 q x 32 keys), fp32 accum (1/sqrt(C) in Qg)
    v8f s[2];
#pragma unroll
    for (int g = 0; g < 2; ++g)
#pragma unroll
      for (int j = 0; j < 8; ++j) s[g][j] = 0.0f;

#pragma unroll
    for (int g = 0; g < 2; ++g) {
#pragma unroll
      for (int kc = 0; kc < 2; ++kc) {
        AF kb;  // B layout: lane col = key (l16); rows = channels lhalf*8+{0..7}, 16+lhalf*8+{0..7}
        const __bf16* kp = Ksh + (g * 16 + l16) * CC + kc * 32 + lhalf * 8;
        kb.u[0] = *(const uint4*)(kp);
        kb.u[1] = *(const uint4*)(kp + 16);
        s[g] = __builtin_amdgcn_wmma_f32_16x16x32_bf16(
            false, qa[kc].v, false, kb.v, (short)0, s[g], false, false);
      }
    }

    // --- row max: 8 independent 16-lane butterflies, step-outer for DS ILP
    float tm[8], al[8];
#pragma unroll
    for (int j = 0; j < 8; ++j) tm[j] = fmaxf(s[0][j], s[1][j]);
#define IPA_RED(MASKV)                                          \
  {                                                             \
    float tt[8];                                                \
    _Pragma("unroll") for (int j = 0; j < 8; ++j)               \
        tt[j] = lane_xor<MASKV>(tm[j]);                         \
    _Pragma("unroll") for (int j = 0; j < 8; ++j)               \
        tm[j] = fmaxf(tm[j], tt[j]);                            \
  }
    IPA_RED(1)
    IPA_RED(2)
    IPA_RED(4)
    IPA_RED(8)
#undef IPA_RED

    // --- rescale + exponentiate + write P tile (bf16) to per-wave LDS
    __bf16* pb = Pbuf[wave];
#pragma unroll
    for (int j = 0; j < 8; ++j) {
      const float mn = fmaxf(mrow[j], tm[j]);
      al[j] = __expf(mrow[j] - mn);
      mrow[j] = mn;
      const float p0 = __expf(s[0][j] - mn);
      const float p1 = __expf(s[1][j] - mn);
      const int m = j + 8 * lhalf;
      pb[m * 32 + l16]      = (__bf16)p0;
      pb[m * 32 + 16 + l16] = (__bf16)p1;
#pragma unroll
      for (int t = 0; t < 4; ++t) oacc[t][j] *= al[j];
    }

    // --- P as A fragment (LDS transpose; same-wave DS ops are in order)
    AF pf;
    pf.u[0] = *(const uint4*)(pb + l16 * 32 + lhalf * 8);
    pf.u[1] = *(const uint4*)(pb + l16 * 32 + 16 + lhalf * 8);

    // --- row sum of P via WMMA with all-ones B: lsum[j] = sum_k P[m][k],
    //     replicated across lanes exactly as the l-state needs.
    v8f zc;
#pragma unroll
    for (int j = 0; j < 8; ++j) zc[j] = 0.0f;
    v8f lsum = __builtin_amdgcn_wmma_f32_16x16x32_bf16(
        false, pf.v, false, ones.v, (short)0, zc, false, false);
#pragma unroll
    for (int j = 0; j < 8; ++j) lrow[j] = lrow[j] * al[j] + lsum[j];

    // --- O += P * V  (16 q x 64 ch), 4 channel tiles
#pragma unroll
    for (int t = 0; t < 4; ++t) {
      AF vb;  // B layout: lane col = channel t*16+l16; rows = keys lhalf*8+{0..7}, 16+lhalf*8+{0..7}
      const __bf16* vp = Vsh + (t * 16 + l16) * 32 + lhalf * 8;
      vb.u[0] = *(const uint4*)(vp);
      vb.u[1] = *(const uint4*)(vp + 16);
      oacc[t] = __builtin_amdgcn_wmma_f32_16x16x32_bf16(
          false, pf.v, false, vb.v, (short)0, oacc[t], false, false);
    }
  }

  // --- normalize, stage to LDS
  float* ob = Obuf[wave];
#pragma unroll
  for (int j = 0; j < 8; ++j) {
    const float inv_l = 1.0f / lrow[j];
    const int m = j + 8 * lhalf;
#pragma unroll
    for (int t = 0; t < 4; ++t) ob[m * CC + t * 16 + l16] = oacc[t][j] * inv_l;
  }

  // --- epilogue: rotate output by lframes[b, q] and write fp32.
  // Lane l16 owns query row q0+l16; lane-halves split the channel work.
  const int qrow = q0 + l16;
  const float* M = lframes + ((size_t)b * NN + qrow) * 16;
  float* orow = out + ((size_t)bh * NN + qrow) * CC;
  const float* ov = ob + l16 * CC;

  if (lhalf == 0) {
#pragma unroll
    for (int c = 0; c < NSC; ++c) orow[c] = ov[c];
  }
  const int vstart = lhalf * 6;
#pragma unroll
  for (int vv = 0; vv < 6; ++vv) {
    const int base = NSC + (vstart + vv) * 4;
    const float x0 = ov[base + 0], x1 = ov[base + 1], x2 = ov[base + 2], x3 = ov[base + 3];
#pragma unroll
    for (int i = 0; i < 4; ++i)
      orow[base + i] = M[i * 4 + 0] * x0 + M[i * 4 + 1] * x1 + M[i * 4 + 2] * x2 + M[i * 4 + 3] * x3;
  }
}

// ---------------------------------------------------------------------------
extern "C" void kernel_launch(void* const* d_in, const int* in_sizes, int n_in,
                              void* d_out, int out_size, void* d_ws, size_t ws_size,
                              hipStream_t stream) {
  const float* q  = (const float*)d_in[0];
  const float* k  = (const float*)d_in[1];
  const float* v  = (const float*)d_in[2];
  const float* lf = (const float*)d_in[3];
  float* out = (float*)d_out;

  // workspace layout
  constexpr size_t BN16 = (size_t)BB * NN * 16;        // 262144 floats
  constexpr size_t E    = (size_t)BB * HH * NN * CC;   // 8388608 elems
  float* invA = (float*)d_ws;
  float* invL = invA + BN16;
  __bf16* Qg = (__bf16*)(invL + BN16);
  __bf16* Kg = Qg + E;
  __bf16* Vt = Kg + E;

  ipa_frames<<<dim3((BB * NN) / 256), dim3(256), 0, stream>>>(lf, invA, invL);
  ipa_transform<<<dim3((BB * HH * NN) / 256), dim3(256), 0, stream>>>(
      q, k, v, invA, invL, Qg, Kg, Vt);
  ipa_attn<<<dim3(NN / 128, BB * HH), dim3(256), 0, stream>>>(Qg, Kg, Vt, lf, out);
}
